// PtrNet1_34677565948674
// MI455X (gfx1250) — matrix operand order repaired and checked by reading, not compiled, not run on hardware
//
#include <hip/hip_runtime.h>
#include <math.h>

typedef __bf16 bf16_t;
typedef __attribute__((ext_vector_type(16))) __bf16 v16bf;
typedef __attribute__((ext_vector_type(8)))  float  v8f;
typedef __attribute__((ext_vector_type(4)))  unsigned int u32x4;

#define BB   512
#define NN   102
#define PP   12
#define HH   256
#define MM   4
#define EE   3
#define NHOPS 2
#define NR   112          // row-padded N (7 x 16 tiles)
#define NK   128          // K-padded N for adjacency GEMM
#define HB   (128*256)    // per-batch padded h buffer elems
#define NEGV (-1e9f)
#define CDIV(a,b) (((a)+(b)-1)/(b))

// ---------------- WMMA fragment helpers (bf16, 16x16x32) ----------------
// A (16xK row-major): lane l -> row l&15; VGPR0..3 = K (l>>4)*8 + [0..7],
//                     VGPR4..7 = K 16 + (l>>4)*8 + [0..7]  (two 16B loads)
static __device__ __forceinline__ v16bf load_a_frag(const bf16_t* __restrict__ A,
                                                    int lda, int row0, int k0) {
  int lane = threadIdx.x & 31;
  int m = lane & 15, g = lane >> 4;
  const bf16_t* p0 = A + (size_t)(row0 + m) * lda + (k0 + g * 8);
  v16bf v;
  u32x4* dv = reinterpret_cast<u32x4*>(&v);
  dv[0] = *reinterpret_cast<const u32x4*>(p0);
  dv[1] = *reinterpret_cast<const u32x4*>(p0 + 16);
  return v;
}
// B as Bt (NxK row-major, i.e. column of B is contiguous): lane l -> col l&15,
// K = (l>>4)*8 + [0..7], 16 + (l>>4)*8 + [0..7] (two 16B loads)
static __device__ __forceinline__ v16bf load_b_frag(const bf16_t* __restrict__ Bt,
                                                    int ldb, int col0, int k0) {
  int lane = threadIdx.x & 31;
  int n = lane & 15, g = lane >> 4;
  const bf16_t* p = Bt + (size_t)(col0 + n) * ldb + (k0 + g * 8);
  v16bf v;
  u32x4* dv = reinterpret_cast<u32x4*>(&v);
  dv[0] = *reinterpret_cast<const u32x4*>(p);
  dv[1] = *reinterpret_cast<const u32x4*>(p + 16);
  return v;
}
static __device__ __forceinline__ v8f wmma_bf16(v16bf a, v16bf b, v8f c) {
  return __builtin_amdgcn_wmma_f32_16x16x32_bf16(false, a, false, b, (short)0, c,
                                                 false, false);
}
static __device__ __forceinline__ v8f acc_zero() {
  v8f a;
#pragma unroll
  for (int i = 0; i < 8; ++i) a[i] = 0.f;
  return a;
}
// async copy of one 16B chunk: global -> LDS (tracked by ASYNCcnt)
static __device__ __forceinline__ void async_copy_b128(bf16_t* lds_p,
                                                       const bf16_t* g_p) {
  unsigned int lva = (unsigned int)(uintptr_t)lds_p;
  asm volatile("global_load_async_to_lds_b128 %0, %1, off"
               :: "v"(lva), "v"(g_p) : "memory");
}
static __device__ __forceinline__ void wait_async0() {
  asm volatile("s_wait_asynccnt 0x0" ::: "memory");
}

// ---------------- prep kernels ----------------
// transpose+convert: src = nmats x [K][Nc] f32  ->  dst = nmats x [Nc][K] bf16
__global__ void k_transpose(const float* __restrict__ src, bf16_t* __restrict__ dst,
                            int K, int Nc, int total) {
  int idx = blockIdx.x * blockDim.x + threadIdx.x;
  if (idx >= total) return;
  int per = K * Nc;
  int mt = idx / per, rem = idx - mt * per;
  int k = rem / Nc, n = rem - k * Nc;
  dst[(size_t)mt * per + (size_t)n * K + k] = (bf16_t)src[idx];
}

__global__ void k_vsum(const float* __restrict__ Vec, float* __restrict__ vsum) {
  int idx = blockIdx.x * blockDim.x + threadIdx.x;           // m*256+h
  if (idx >= MM * HH) return;
  const float* p = Vec + (size_t)idx * HH;
  float s = 0.f;
#pragma unroll 4
  for (int k = 0; k < HH; ++k) s += p[k];
  vsum[idx] = s;
}

__global__ void k_constq(const float* __restrict__ W_q, const float* __restrict__ dec,
                         const float* __restrict__ v1, float* __restrict__ cq) {
  int idx = blockIdx.x * blockDim.x + threadIdx.x;           // m*256+h
  if (idx >= MM * HH) return;
  int m = idx >> 8, h = idx & 255;
  const float* Wm = W_q + (size_t)m * 768 * 256;
  float s = 0.f;
  for (int j = 0; j < 256; ++j)
    s += dec[j] * Wm[(size_t)(256 + j) * 256 + h] + v1[j] * Wm[(size_t)(512 + j) * 256 + h];
  cq[idx] = s;
}

// embedding: h0 = nf @ W_emb, write padded bf16 h and hT
__global__ void k_embed(const float* __restrict__ nf, const float* __restrict__ W_emb,
                        bf16_t* __restrict__ h, bf16_t* __restrict__ hT) {
  int idx = blockIdx.x * blockDim.x + threadIdx.x;
  if (idx >= BB * 128 * HH) return;
  int hc = idx & 255, r = (idx >> 8) & 127, b = idx >> 15;
  float acc = 0.f;
  if (r < NN) {
    const float* x = nf + ((size_t)b * NN + r) * PP;
#pragma unroll
    for (int p = 0; p < PP; ++p) acc += x[p] * W_emb[p * HH + hc];
  }
  bf16_t v = (bf16_t)acc;
  h[(size_t)b * HB + r * 256 + hc] = v;
  hT[(size_t)b * HB + hc * 128 + r] = v;
}

__global__ void k_pack_adj(const float* __restrict__ adj, bf16_t* __restrict__ adjp) {
  int idx = blockIdx.x * blockDim.x + threadIdx.x;
  if (idx >= BB * EE * NR * NK) return;
  int col = idx & 127, row = (idx >> 7) % NR, be = idx / (NR * NK);
  float v = 0.f;
  if (row < NN && col < NN) v = adj[((size_t)be * NN + row) * NN + col];
  adjp[idx] = (bf16_t)v;
}

// ---------------- WMMA GEMM kernels ----------------
// msg[b,e] (112x256) = adjp[b,e] (112x128) @ h[b] (128x256).
// One wave = 16 rows x 64 cols (4 accumulators, A fragment reused 4x).
__global__ void __launch_bounds__(32) k_msg(const bf16_t* __restrict__ adjp,
                                            const bf16_t* __restrict__ hT,
                                            bf16_t* __restrict__ msg) {
  int rt = blockIdx.x, cq4 = blockIdx.y, be = blockIdx.z;
  int b = be / EE;
  const bf16_t* A  = adjp + (size_t)be * NR * NK;
  const bf16_t* Bt = hT + (size_t)b * HB;
  int row0 = rt * 16, col0 = cq4 * 64;
  v8f acc[4];
#pragma unroll
  for (int cc = 0; cc < 4; ++cc) acc[cc] = acc_zero();
#pragma unroll
  for (int k0 = 0; k0 < NK; k0 += 32) {
    if (k0 + 32 < NK)
      __builtin_prefetch(A + (size_t)row0 * NK + k0 + 32, 0, 1);
    v16bf a = load_a_frag(A, NK, row0, k0);
#pragma unroll
    for (int cc = 0; cc < 4; ++cc)
      acc[cc] = wmma_bf16(a, load_b_frag(Bt, 128, col0 + cc * 16, k0), acc[cc]);
  }
  int lane = threadIdx.x & 31;
  int rbase = row0 + ((lane >> 4) << 3);
  bf16_t* out = msg + (size_t)be * NR * HH;
#pragma unroll
  for (int cc = 0; cc < 4; ++cc) {
    int coln = col0 + cc * 16 + (lane & 15);
#pragma unroll
    for (int v = 0; v < 8; ++v)
      out[(size_t)(rbase + v) * HH + coln] = (bf16_t)acc[cc][v];
  }
}

// h_out = relu( sum_e msg[b,e] @ W_rel[e] + h_in @ W_self ).
// Block = 4 waves (128 thr). The four 16x256 A panels (hin + 3 msg) are staged
// into LDS once via global_load_async_to_lds_b128 and shared by all waves;
// wave w computes cols [w*64, w*64+64) => block covers a full 16x256 row tile.
__global__ void __launch_bounds__(128) k_combine(const bf16_t* __restrict__ msg,
                                                 const bf16_t* __restrict__ hin,
                                                 const bf16_t* __restrict__ WrelT,
                                                 const bf16_t* __restrict__ WselfT,
                                                 bf16_t* __restrict__ hout,
                                                 bf16_t* __restrict__ houtT) {
  __shared__ bf16_t Asmem[4 * 16 * 256];   // 32 KB
  int tid = threadIdx.x;
  int rt = blockIdx.x, b = blockIdx.y;
  int row0 = rt * 16;

  const bf16_t* srcs[4];
  srcs[0] = hin + (size_t)b * HB + (size_t)row0 * 256;
  srcs[1] = msg + ((size_t)b * EE + 0) * NR * HH + (size_t)row0 * 256;
  srcs[2] = msg + ((size_t)b * EE + 1) * NR * HH + (size_t)row0 * 256;
  srcs[3] = msg + ((size_t)b * EE + 2) * NR * HH + (size_t)row0 * 256;
  // 4 mats * 16 rows * 32 chunks(16B) = 2048 chunks; 16 per thread
#pragma unroll
  for (int it = 0; it < 16; ++it) {
    int c = tid + it * 128;
    int mat = c >> 9, rem = c & 511, r = rem >> 5, ch = rem & 31;
    async_copy_b128(&Asmem[(mat * 16 + r) * 256 + ch * 8], srcs[mat] + r * 256 + ch * 8);
  }
  wait_async0();
  __syncthreads();

  int w = tid >> 5;            // wave id -> 64-col slice
  int lane = tid & 31;
  int mrow = lane & 15, g2 = lane >> 4;
  v8f acc[4];
#pragma unroll
  for (int cc = 0; cc < 4; ++cc) acc[cc] = acc_zero();
#pragma unroll
  for (int mat = 0; mat < 4; ++mat) {
    const bf16_t* Wt = (mat == 0) ? WselfT : (WrelT + (size_t)(mat - 1) * HH * HH);
#pragma unroll
    for (int k0 = 0; k0 < HH; k0 += 32) {
      const bf16_t* p0 = &Asmem[mat * 4096 + mrow * 256 + k0 + g2 * 8];
      v16bf a;
      u32x4* dv = reinterpret_cast<u32x4*>(&a);
      dv[0] = *reinterpret_cast<const u32x4*>(p0);
      dv[1] = *reinterpret_cast<const u32x4*>(p0 + 16);
#pragma unroll
      for (int cc = 0; cc < 4; ++cc)
        acc[cc] = wmma_bf16(a, load_b_frag(Wt, HH, w * 64 + cc * 16, k0), acc[cc]);
    }
  }
  int rbase = row0 + (g2 << 3);
#pragma unroll
  for (int cc = 0; cc < 4; ++cc) {
    int coln = w * 64 + cc * 16 + (lane & 15);
#pragma unroll
    for (int v = 0; v < 8; ++v) {
      float x = acc[cc][v];
      x = x > 0.f ? x : 0.f;
      bf16_t hv = (bf16_t)x;
      int row = rbase + v;
      hout[(size_t)b * HB + row * 256 + coln]  = hv;
      houtT[(size_t)b * HB + coln * 128 + row] = hv;
    }
  }
}

__global__ void k_zero_tail(bf16_t* __restrict__ h, bf16_t* __restrict__ hT) {
  int idx = blockIdx.x * blockDim.x + threadIdx.x;
  if (idx >= BB * 16 * HH) return;
  int hc = idx & 255, rr = NR + ((idx >> 8) & 15), b = idx >> 12;
  h[(size_t)b * HB + rr * 256 + hc]  = (bf16_t)0.f;
  hT[(size_t)b * HB + hc * 128 + rr] = (bf16_t)0.f;
}

// R[b] (112x256 f32) = enc[b] (112x256) @ Wt^T. Wave = 16 rows x 64 cols.
__global__ void __launch_bounds__(32) k_refgemm(const bf16_t* __restrict__ enc,
                                                const bf16_t* __restrict__ Wt,
                                                float* __restrict__ R) {
  int rt = blockIdx.x, cq4 = blockIdx.y, b = blockIdx.z;
  const bf16_t* A = enc + (size_t)b * HB;
  int row0 = rt * 16, col0 = cq4 * 64;
  v8f acc[4];
#pragma unroll
  for (int cc = 0; cc < 4; ++cc) acc[cc] = acc_zero();
#pragma unroll
  for (int k0 = 0; k0 < HH; k0 += 32) {
    if (k0 + 32 < HH)
      __builtin_prefetch(A + (size_t)row0 * HH + k0 + 32, 0, 1);
    v16bf a = load_a_frag(A, HH, row0, k0);
#pragma unroll
    for (int cc = 0; cc < 4; ++cc)
      acc[cc] = wmma_bf16(a, load_b_frag(Wt, HH, col0 + cc * 16, k0), acc[cc]);
  }
  int lane = threadIdx.x & 31;
  int rbase = row0 + ((lane >> 4) << 3);
  float* out = R + (size_t)b * NR * HH;
#pragma unroll
  for (int cc = 0; cc < 4; ++cc) {
    int coln = col0 + cc * 16 + (lane & 15);
#pragma unroll
    for (int v = 0; v < 8; ++v)
      out[(size_t)(rbase + v) * HH + coln] = acc[cc][v];
  }
}

// ---------------- decoder small kernels ----------------
__global__ void k_mean(const bf16_t* __restrict__ encT, float* __restrict__ em) {
  int idx = blockIdx.x * blockDim.x + threadIdx.x;          // b*256+h
  if (idx >= BB * HH) return;
  int b = idx >> 8, h = idx & 255;
  const bf16_t* p = encT + (size_t)b * HB + h * 128;
  float s = 0.f;
#pragma unroll 4
  for (int n = 0; n < NN; ++n) s += (float)p[n];
  em[idx] = s * (1.f / (float)NN);
}

__global__ void k_q(const float* __restrict__ em, const float* __restrict__ W_q,
                    const float* __restrict__ cq, float* __restrict__ q) {
  int idx = blockIdx.x * blockDim.x + threadIdx.x;          // b*1024 + m*256 + h
  if (idx >= BB * MM * HH) return;
  int h = idx & 255, m = (idx >> 8) & 3, b = idx >> 10;
  const float* e  = em + (size_t)b * HH;
  const float* Wm = W_q + (size_t)m * 768 * 256;
  float s = cq[m * HH + h];
  for (int f = 0; f < HH; ++f) s += e[f] * Wm[(size_t)f * 256 + h];
  q[idx] = s;
}

// scores[b,m,n] = sum_h tanh(q[b,m,h] + R[b,n,h]) * vsum[m,h]
__global__ void k_scores(const float* __restrict__ q, const float* __restrict__ R,
                         const float* __restrict__ vsum, float* __restrict__ scores, int m) {
  int idx = blockIdx.x * blockDim.x + threadIdx.x;
  if (idx >= BB * NN) return;
  int b = idx / NN, n = idx - b * NN;
  const float* qb = q + ((size_t)b * MM + m) * HH;
  const float* Rr = R + ((size_t)b * NR + n) * HH;
  const float* vs = vsum + m * HH;
  float s = 0.f;
#pragma unroll 4
  for (int h = 0; h < HH; ++h) s += tanhf(qb[h] + Rr[h]) * vs[h];
  scores[((size_t)b * MM + m) * NR + n] = s;
}

// per (b,m): masked softmax over n, then ctx[b,m,h] = sum_n a_n enc[b,n,h]
__global__ void k_softmax_ctx(const float* __restrict__ scores, const int* __restrict__ mask,
                              const bf16_t* __restrict__ encT, float* __restrict__ ctx) {
  __shared__ float sa[NN];
  __shared__ float sw[NN];
  __shared__ float red[128];
  int bm = blockIdx.x, b = bm >> 2, m = bm & 3;
  int tid = threadIdx.x;
  if (tid < NN) {
    float s = scores[(size_t)bm * NR + tid];
    if (mask[b * NN + tid] > 0) s = NEGV;
    sa[tid] = s;
  }
  __syncthreads();
  red[tid] = (tid < NN) ? sa[tid] : -3.4e38f;
  __syncthreads();
  for (int s2 = 64; s2 > 0; s2 >>= 1) {
    if (tid < s2) red[tid] = fmaxf(red[tid], red[tid + s2]);
    __syncthreads();
  }
  float mx = red[0];
  __syncthreads();
  float e = (tid < NN) ? expf(sa[tid] - mx) : 0.f;
  if (tid < NN) sw[tid] = e;
  red[tid] = e;
  __syncthreads();
  for (int s2 = 64; s2 > 0; s2 >>= 1) {
    if (tid < s2) red[tid] += red[tid + s2];
    __syncthreads();
  }
  float inv = 1.f / red[0];
  for (int h = tid; h < HH; h += 128) {
    const bf16_t* p = encT + (size_t)b * HB + h * 128;
    float acc = 0.f;
#pragma unroll 4
    for (int n = 0; n < NN; ++n) acc += sw[n] * (float)p[n];
    ctx[(size_t)b * (MM * HH) + m * HH + h] = acc * inv;
  }
}

__global__ void k_query2(const float* __restrict__ ctx, const float* __restrict__ W_mh,
                         float* __restrict__ q2) {
  int idx = blockIdx.x * blockDim.x + threadIdx.x;          // b*256+h
  if (idx >= BB * HH) return;
  int h = idx & 255, b = idx >> 8;
  const float* c = ctx + (size_t)b * (MM * HH);
  float s = 0.f;
  for (int f = 0; f < MM * HH; ++f) s += c[f] * W_mh[(size_t)f * 256 + h];
  q2[idx] = s;
}

__global__ void k_q2w(const float* __restrict__ q2, const float* __restrict__ W_q2,
                      float* __restrict__ q2w) {
  int idx = blockIdx.x * blockDim.x + threadIdx.x;          // b*256+h
  if (idx >= BB * HH) return;
  int h = idx & 255, b = idx >> 8;
  const float* qb = q2 + (size_t)b * HH;
  float s = 0.f;
  for (int j = 0; j < HH; ++j) s += qb[j] * W_q2[(size_t)j * 256 + h];
  q2w[idx] = s;
}

// final: u2 = 10*tanh(Vec2 . tanh(q2w + R2)); mask; log_softmax over n
__global__ void k_final(const float* __restrict__ R, const float* __restrict__ q2w,
                        const float* __restrict__ Vec2, const int* __restrict__ mask,
                        float* __restrict__ out) {
  __shared__ float su[NN];
  __shared__ float red[128];
  int b = blockIdx.x, tid = threadIdx.x;
  if (tid < NN) {
    const float* Rr = R + ((size_t)b * NR + tid) * HH;
    const float* qb = q2w + (size_t)b * HH;
    float s = 0.f;
#pragma unroll 4
    for (int h = 0; h < HH; ++h) s += Vec2[h] * tanhf(qb[h] + Rr[h]);
    s = 10.f * tanhf(s);
    if (mask[b * NN + tid] > 0) s = NEGV;
    su[tid] = s;
  }
  __syncthreads();
  red[tid] = (tid < NN) ? su[tid] : -3.4e38f;
  __syncthreads();
  for (int s2 = 64; s2 > 0; s2 >>= 1) {
    if (tid < s2) red[tid] = fmaxf(red[tid], red[tid + s2]);
    __syncthreads();
  }
  float mx = red[0];
  __syncthreads();
  red[tid] = (tid < NN) ? expf(su[tid] - mx) : 0.f;
  __syncthreads();
  for (int s2 = 64; s2 > 0; s2 >>= 1) {
    if (tid < s2) red[tid] += red[tid + s2];
    __syncthreads();
  }
  float lse = logf(red[0]);
  if (tid < NN) out[(size_t)b * NN + tid] = su[tid] - mx - lse;
}

// ---------------- host ----------------
extern "C" void kernel_launch(void* const* d_in, const int* in_sizes, int n_in,
                              void* d_out, int out_size, void* d_ws, size_t ws_size,
                              hipStream_t stream) {
  const float* nf     = (const float*)d_in[0];
  const float* adj    = (const float*)d_in[1];
  const int*   mask   = (const int*)d_in[2];
  const float* W_emb  = (const float*)d_in[3];
  const float* W_rel  = (const float*)d_in[4];
  const float* W_self = (const float*)d_in[5];
  const float* Vec    = (const float*)d_in[6];
  const float* W_q    = (const float*)d_in[7];
  const float* W_ref  = (const float*)d_in[8];
  const float* W_mh   = (const float*)d_in[9];
  const float* Vec2   = (const float*)d_in[10];
  const float* W_q2   = (const float*)d_in[11];
  const float* W_ref2 = (const float*)d_in[12];
  const float* dec    = (const float*)d_in[13];
  const float* v1     = (const float*)d_in[14];
  float* out = (float*)d_out;

  char* wsb = (char*)d_ws;
  size_t off = 0;
  auto take = [&](size_t bytes) -> char* {
    char* p = wsb + off;
    off = (off + bytes + 255) & ~(size_t)255;
    return p;
  };
  bf16_t* adjp  = (bf16_t*)take((size_t)BB * EE * NR * NK * 2);
  bf16_t* hp0   = (bf16_t*)take((size_t)BB * HB * 2);
  bf16_t* hp0T  = (bf16_t*)take((size_t)BB * HB * 2);
  bf16_t* hp1   = (bf16_t*)take((size_t)BB * HB * 2);
  bf16_t* hp1T  = (bf16_t*)take((size_t)BB * HB * 2);
  bf16_t* msgb  = (bf16_t*)take((size_t)BB * EE * NR * HH * 2);
  float*  Rbuf  = (float*)take((size_t)BB * NR * HH * 4);
  bf16_t* WrelT = (bf16_t*)take((size_t)NHOPS * EE * HH * HH * 2);
  bf16_t* WselfT= (bf16_t*)take((size_t)NHOPS * HH * HH * 2);
  bf16_t* WrefT = (bf16_t*)take((size_t)MM * HH * HH * 2);
  bf16_t* Wref2T= (bf16_t*)take((size_t)HH * HH * 2);
  float*  emean = (float*)take((size_t)BB * HH * 4);
  float*  cq    = (float*)take((size_t)MM * HH * 4);
  float*  qbuf  = (float*)take((size_t)BB * MM * HH * 4);
  float*  vsum  = (float*)take((size_t)MM * HH * 4);
  float*  scoresb=(float*)take((size_t)BB * MM * NR * 4);
  float*  ctx   = (float*)take((size_t)BB * MM * HH * 4);
  float*  q2    = (float*)take((size_t)BB * HH * 4);
  float*  q2w   = (float*)take((size_t)BB * HH * 4);
  (void)ws_size; (void)n_in; (void)in_sizes; (void)out_size;

  // ---- weight prep ----
  k_transpose<<<CDIV(NHOPS * EE * HH * HH, 256), 256, 0, stream>>>(W_rel, WrelT, HH, HH, NHOPS * EE * HH * HH);
  k_transpose<<<CDIV(NHOPS * HH * HH, 256), 256, 0, stream>>>(W_self, WselfT, HH, HH, NHOPS * HH * HH);
  k_transpose<<<CDIV(MM * HH * HH, 256), 256, 0, stream>>>(W_ref, WrefT, HH, HH, MM * HH * HH);
  k_transpose<<<CDIV(HH * HH, 256), 256, 0, stream>>>(W_ref2, Wref2T, HH, HH, HH * HH);
  k_vsum<<<CDIV(MM * HH, 256), 256, 0, stream>>>(Vec, vsum);
  k_constq<<<CDIV(MM * HH, 256), 256, 0, stream>>>(W_q, dec, v1, cq);

  // ---- encoder ----
  k_embed<<<CDIV(BB * 128 * HH, 256), 256, 0, stream>>>(nf, W_emb, hp0, hp0T);
  k_pack_adj<<<CDIV(BB * EE * NR * NK, 256), 256, 0, stream>>>(adj, adjp);

  const bf16_t* hin = hp0;
  const bf16_t* hinT = hp0T;
  bf16_t* houtArr[2]  = {hp1, hp0};
  bf16_t* houtTArr[2] = {hp1T, hp0T};
  for (int hop = 0; hop < NHOPS; ++hop) {
    bf16_t* hout  = houtArr[hop & 1];
    bf16_t* houtT = houtTArr[hop & 1];
    k_msg<<<dim3(NR / 16, HH / 64, BB * EE), 32, 0, stream>>>(adjp, hinT, msgb);
    k_combine<<<dim3(NR / 16, BB), 128, 0, stream>>>(
        msgb, hin, WrelT + (size_t)hop * EE * HH * HH, WselfT + (size_t)hop * HH * HH,
        hout, houtT);
    k_zero_tail<<<CDIV(BB * 16 * HH, 256), 256, 0, stream>>>(hout, houtT);
    hin = hout;
    hinT = houtT;
  }
  const bf16_t* enc  = hin;   // == hp0 after 2 hops
  const bf16_t* encT = hinT;

  // ---- decoder ----
  k_mean<<<CDIV(BB * HH, 256), 256, 0, stream>>>(encT, emean);
  k_q<<<CDIV(BB * MM * HH, 256), 256, 0, stream>>>(emean, W_q, cq, qbuf);
  for (int m = 0; m < MM; ++m) {
    k_refgemm<<<dim3(NR / 16, HH / 64, BB), 32, 0, stream>>>(enc, WrefT + (size_t)m * HH * HH, Rbuf);
    k_scores<<<CDIV(BB * NN, 256), 256, 0, stream>>>(qbuf, Rbuf, vsum, scoresb, m);
  }
  k_softmax_ctx<<<BB * MM, 128, 0, stream>>>(scoresb, mask, encT, ctx);
  k_query2<<<CDIV(BB * HH, 256), 256, 0, stream>>>(ctx, W_mh, q2);
  k_q2w<<<CDIV(BB * HH, 256), 256, 0, stream>>>(q2, W_q2, q2w);
  k_refgemm<<<dim3(NR / 16, HH / 64, BB), 32, 0, stream>>>(enc, Wref2T, Rbuf);
  k_final<<<BB, 128, 0, stream>>>(Rbuf, q2w, Vec2, mask, out);
}
